// Span_selection_model_en_89678917141007
// MI455X (gfx1250) — matrix operand order repaired and checked by reading, not compiled, not run on hardware
//
#include <hip/hip_runtime.h>
#include <hip/hip_bf16.h>

// ---------------------------------------------------------------------------
// Span_selection_model for MI455X (gfx1250, wave32, WMMA).
// All GEMM-like stages use v_wmma_f32_16x16x32_f16 (fp32 -> f16 fragments).
// Branchless fragment loads (index clamping + zero-padded K), float4 vector
// loads, K-loop unrolled x2, template-specialized epilogues.
// ---------------------------------------------------------------------------

typedef __attribute__((ext_vector_type(16))) _Float16 v16h;
typedef __attribute__((ext_vector_type(8)))  float    v8f;

namespace cfg {
constexpr int B  = 2;
constexpr int S  = 64;
constexpr int H  = 768;
constexpr int W  = 6;
constexpr int NH = 12;
constexpr int DH = 64;
constexpr int R  = 12;
constexpr int TAGS = 2;
constexpr int L  = 369;          // S*W - W*(W-1)/2
constexpr int LP = 384;          // L padded to multiple of 32 (K-dim padding)
constexpr int H4 = 4 * H;        // 3072
constexpr int H2 = 2 * H;        // 1536
}
using namespace cfg;

// -------------------------- WMMA helpers -----------------------------------

__device__ __forceinline__ v8f wmma32(v16h a, v16h b, v8f c) {
  return __builtin_amdgcn_wmma_f32_16x16x32_f16(false, a, false, b, (short)0, c,
                                                false, false);
}

__device__ __forceinline__ void cvt16(v16h& r, float4 x0, float4 x1, float4 x2,
                                      float4 x3) {
  r[0] = (_Float16)x0.x; r[1] = (_Float16)x0.y; r[2] = (_Float16)x0.z; r[3] = (_Float16)x0.w;
  r[4] = (_Float16)x1.x; r[5] = (_Float16)x1.y; r[6] = (_Float16)x1.z; r[7] = (_Float16)x1.w;
  r[8] = (_Float16)x2.x; r[9] = (_Float16)x2.y; r[10] = (_Float16)x2.z; r[11] = (_Float16)x2.w;
  r[12] = (_Float16)x3.x; r[13] = (_Float16)x3.y; r[14] = (_Float16)x3.z; r[15] = (_Float16)x3.w;
}

// A fragment 16x32 f16 from row-major fp32 A[m][k]. Branchless: row clamped
// (out-of-range rows only feed unstored output rows). K multiple of 32 with
// zeroed padding. M = lane&15, kb=(lane>>4)*16; k=kb+e.
__device__ __forceinline__ v16h frag_a(const float* A, int lda, int m0, int k0,
                                       int Mmax) {
  const int lane = threadIdx.x & 31;
  int m = m0 + (lane & 15);
  m = (m < Mmax) ? m : (Mmax - 1);
  const int kb = (lane >> 4) * 16;
  const float4* p = reinterpret_cast<const float4*>(A + m * lda + k0 + kb);
  v16h r;
  cvt16(r, p[0], p[1], p[2], p[3]);   // +0..15 floats
  return r;
}

// B fragment 32x16 f16 from weights stored [n][k] (computing X @ W^T).
// Contiguous k order (matches frag_a): N = lane&15, K = (lane>>4)*16 + e. Branchless, n clamped.
__device__ __forceinline__ v16h frag_b_nk(const float* Wt, int ldw, int n0,
                                          int k0, int Nmax) {
  const int lane = threadIdx.x & 31;
  int n = n0 + (lane & 15);
  n = (n < Nmax) ? n : (Nmax - 1);
  const int kb = (lane >> 4) * 16;
  const float4* p = reinterpret_cast<const float4*>(Wt + n * ldw + k0 + kb);
  v16h r;
  cvt16(r, p[0], p[1], p[2], p[3]);
  return r;
}

__device__ __forceinline__ float sigf(float x) {
  return 1.0f / (1.0f + __expf(-x));
}

// -------------------------- Stage 1: spans + concat -------------------------

__device__ __forceinline__ void span_iw(int l, int& i, int& w) {
  if (l < (S - W + 1) * W) { i = l / W; w = (l % W) + 1; return; }
  int r = l - (S - W + 1) * W;
  i = S - W + 1;
  int cap = W - 1;
  while (r >= cap) { r -= cap; --cap; ++i; }
  w = r + 1;
}

__global__ void spans_kernel(const float* __restrict__ seq,
                             const float* __restrict__ pool,
                             float* __restrict__ xin) {
  const int bl = blockIdx.x;              // B*L
  const int b = bl / L, l = bl % L;
  int i, w;
  span_iw(l, i, w);
  const float4* base = reinterpret_cast<const float4*>(seq + (b * S + i) * H);
  const float4* pl   = reinterpret_cast<const float4*>(pool + b * H);
  float4* xo = reinterpret_cast<float4*>(xin + bl * H2);
  for (int h4 = threadIdx.x; h4 < H / 4; h4 += blockDim.x) {
    float4 m = base[h4];
    for (int ww = 1; ww < w; ++ww) {
      float4 s = base[ww * (H / 4) + h4];
      m.x = fmaxf(m.x, s.x); m.y = fmaxf(m.y, s.y);
      m.z = fmaxf(m.z, s.z); m.w = fmaxf(m.w, s.w);
    }
    xo[h4] = m;
    xo[H / 4 + h4] = pl[h4];
  }
}

// -------------------------- Generic WMMA GEMM ------------------------------
// C[m][n] = act( sum_k A[m][k]*W[n][k] + bias[n] (+ res[m][n]) )
// Template-specialized epilogue; K-loop unrolled x2 (K % 64 == 0).

template <bool HAS_BIAS, bool HAS_RES, bool ACT_TANH>
__global__ void gemm_nk_kernel(const float* __restrict__ A,
                               const float* __restrict__ Wt,
                               const float* __restrict__ bias,
                               const float* __restrict__ res,
                               float* __restrict__ C,
                               int M, int N, int K,
                               int lda, int ldw, int ldc, int ldres) {
  const int wave = threadIdx.x >> 5;
  const int n0 = (blockIdx.x * 4 + wave) * 16;
  const int m0 = blockIdx.y * 16;
  if (n0 >= N) return;                    // wave-uniform
  v8f acc = {};
  for (int k0 = 0; k0 < K; k0 += 64) {
    v16h a0 = frag_a(A, lda, m0, k0, M);
    v16h b0 = frag_b_nk(Wt, ldw, n0, k0, N);
    v16h a1 = frag_a(A, lda, m0, k0 + 32, M);
    v16h b1 = frag_b_nk(Wt, ldw, n0, k0 + 32, N);
    acc = wmma32(a0, b0, acc);
    acc = wmma32(a1, b1, acc);
  }
  const int lane = threadIdx.x & 31;
  const int n = n0 + (lane & 15);
  const int mo = (lane >> 4) * 8;
  if (n < N) {
    float bb = HAS_BIAS ? bias[n] : 0.0f;
#pragma unroll
    for (int r2 = 0; r2 < 8; ++r2) {
      int m = m0 + mo + r2;
      if (m < M) {
        float v = acc[r2] + bb;
        if (HAS_RES) v += res[m * ldres + n];
        if (ACT_TANH) v = tanhf(v);
        C[m * ldc + n] = v;
      }
    }
  }
}

// V projection variant: stores transposed per batch, vT[(b*N+n)*LP + l],
// m = b*L + l. K-pad columns zeroed separately.
__global__ void gemm_nk_t_kernel(const float* __restrict__ A,
                                 const float* __restrict__ Wt,
                                 const float* __restrict__ bias,
                                 float* __restrict__ CT,
                                 int M, int N, int K, int lda, int ldw) {
  const int wave = threadIdx.x >> 5;
  const int n0 = (blockIdx.x * 4 + wave) * 16;
  const int m0 = blockIdx.y * 16;
  if (n0 >= N) return;
  v8f acc = {};
  for (int k0 = 0; k0 < K; k0 += 64) {
    v16h a0 = frag_a(A, lda, m0, k0, M);
    v16h b0 = frag_b_nk(Wt, ldw, n0, k0, N);
    v16h a1 = frag_a(A, lda, m0, k0 + 32, M);
    v16h b1 = frag_b_nk(Wt, ldw, n0, k0 + 32, N);
    acc = wmma32(a0, b0, acc);
    acc = wmma32(a1, b1, acc);
  }
  const int lane = threadIdx.x & 31;
  const int n = n0 + (lane & 15);
  const int mo = (lane >> 4) * 8;
  if (n < N) {
    float bb = bias[n];
#pragma unroll
    for (int r2 = 0; r2 < 8; ++r2) {
      int m = m0 + mo + r2;
      if (m < M) {
        int b2 = m / L, l2 = m - b2 * L;
        CT[(b2 * N + n) * LP + l2] = acc[r2] + bb;
      }
    }
  }
}

// Zero the K-pad columns l in [L, LP) of vT.
__global__ void vt_padzero_kernel(float* __restrict__ vT) {
  const int idx = blockIdx.x * blockDim.x + threadIdx.x;   // B*H*(LP-L)
  const int PAD = LP - L;
  if (idx >= B * H * PAD) return;
  const int row = idx / PAD;
  const int l = L + idx % PAD;
  vT[row * LP + l] = 0.0f;
}

// -------------------------- Stage 3: sequential LSTM ------------------------
// One persistent workgroup (512 threads = 16 waves); h/c/g LDS-resident,
// Wih/Whh stream from L2 through WMMA (M=2 valid rows per 16-tile).

__global__ void __launch_bounds__(512)
lstm_kernel(const float* __restrict__ xc, const float* __restrict__ Wih,
            const float* __restrict__ Whh, const float* __restrict__ bih,
            const float* __restrict__ bhh, float* __restrict__ xo) {
  __shared__ float sh_h[B][H];
  __shared__ float sh_c[B][H];
  __shared__ float sh_g[B][H4];
  const int tid = threadIdx.x;
  for (int idx = tid; idx < B * H; idx += 512) {
    (&sh_h[0][0])[idx] = 0.0f;
    (&sh_c[0][0])[idx] = 0.0f;
  }
  __syncthreads();
  const int wave = tid >> 5;
  const int lane = tid & 31;
  for (int t = 0; t < L; ++t) {
    const float* xt = xc + t * H;               // batch rows stride L*H
    for (int tile = wave; tile < H4 / 16; tile += 16) {
      const int n0 = tile * 16;
      v8f acc = {};
      for (int k0 = 0; k0 < H; k0 += 64) {
        v16h a0 = frag_a(xt, L * H, 0, k0, B);
        v16h b0 = frag_b_nk(Wih, H, n0, k0, H4);
        v16h a1 = frag_a(xt, L * H, 0, k0 + 32, B);
        v16h b1 = frag_b_nk(Wih, H, n0, k0 + 32, H4);
        acc = wmma32(a0, b0, acc);
        acc = wmma32(a1, b1, acc);
      }
      for (int k0 = 0; k0 < H; k0 += 64) {
        v16h a0 = frag_a(&sh_h[0][0], H, 0, k0, B);
        v16h b0 = frag_b_nk(Whh, H, n0, k0, H4);
        v16h a1 = frag_a(&sh_h[0][0], H, 0, k0 + 32, B);
        v16h b1 = frag_b_nk(Whh, H, n0, k0 + 32, H4);
        acc = wmma32(a0, b0, acc);
        acc = wmma32(a1, b1, acc);
      }
      const int n = n0 + (lane & 15);
      const int mo = (lane >> 4) * 8;
#pragma unroll
      for (int r2 = 0; r2 < 8; ++r2) {
        int m = mo + r2;
        if (m < B) sh_g[m][n] = acc[r2];
      }
    }
    __syncthreads();
    for (int idx = tid; idx < B * H; idx += 512) {
      const int b = idx / H, n = idx % H;
      float gi = sh_g[b][n]         + bih[n]         + bhh[n];
      float gf = sh_g[b][H + n]     + bih[H + n]     + bhh[H + n];
      float gg = sh_g[b][2 * H + n] + bih[2 * H + n] + bhh[2 * H + n];
      float go = sh_g[b][3 * H + n] + bih[3 * H + n] + bhh[3 * H + n];
      float cc = sigf(gf) * sh_c[b][n] + sigf(gi) * tanhf(gg);
      float hh = sigf(go) * tanhf(cc);
      sh_c[b][n] = cc;
      sh_h[b][n] = hh;
      xo[(b * L + t) * H + n] = hh;
    }
    __syncthreads();
  }
}

// -------------------------- Stage 5: attention scores -----------------------
// Padded scores buffer: per (b,h) an LP x LP tile; pads written 0 so the
// later P@V K-reduction over LP is exact.

__global__ void scores_kernel(const float* __restrict__ q,
                              const float* __restrict__ k,
                              const float* __restrict__ mask,
                              float* __restrict__ sc) {
  const int wave = threadIdx.x >> 5;
  const int nt = blockIdx.x * 4 + wave;   // key tile 0..23
  const int mt = blockIdx.y;              // query tile 0..23
  const int bh = blockIdx.z;              // 0..B*NH-1
  const int b = bh / NH, h = bh % NH;
  const float* qb = q + b * L * H + h * DH;
  const float* kb = k + b * L * H + h * DH;
  v8f acc = {};
  {
    v16h a0 = frag_a(qb, H, mt * 16, 0, L);
    v16h b0 = frag_b_nk(kb, H, nt * 16, 0, L);
    v16h a1 = frag_a(qb, H, mt * 16, 32, L);
    v16h b1 = frag_b_nk(kb, H, nt * 16, 32, L);
    acc = wmma32(a0, b0, acc);
    acc = wmma32(a1, b1, acc);
  }
  const int lane = threadIdx.x & 31;
  const int n = nt * 16 + (lane & 15);
  const int mo = (lane >> 4) * 8;
  const int nc = (n < L) ? n : (L - 1);
  const float am = (1.0f - mask[b * L + nc]) * -10000.0f;
  float* out = sc + (size_t)bh * LP * LP;
#pragma unroll
  for (int r2 = 0; r2 < 8; ++r2) {
    int m = mt * 16 + mo + r2;
    float v = (m < L && n < L) ? (acc[r2] * 0.125f + am) : 0.0f;
    out[m * LP + n] = v;
  }
}

// -------------------------- Stage 6: row softmax ----------------------------

__global__ void softmax_kernel(float* __restrict__ sc) {
  __shared__ float red[4];
  const int bh = blockIdx.x / L;
  const int l  = blockIdx.x % L;
  float* row = sc + ((size_t)bh * LP + l) * LP;
  const int tid = threadIdx.x;            // 128
  float mx = -1e30f;
  for (int j = tid; j < L; j += 128) mx = fmaxf(mx, row[j]);
  for (int off = 16; off; off >>= 1) mx = fmaxf(mx, __shfl_xor(mx, off, 32));
  if ((tid & 31) == 0) red[tid >> 5] = mx;
  __syncthreads();
  mx = fmaxf(fmaxf(red[0], red[1]), fmaxf(red[2], red[3]));
  __syncthreads();
  float s = 0.0f;
  for (int j = tid; j < L; j += 128) s += __expf(row[j] - mx);
  for (int off = 16; off; off >>= 1) s += __shfl_xor(s, off, 32);
  if ((tid & 31) == 0) red[tid >> 5] = s;
  __syncthreads();
  s = red[0] + red[1] + red[2] + red[3];
  const float inv = 1.0f / s;
  for (int j = tid; j < L; j += 128) row[j] = __expf(row[j] - mx) * inv;
}

// -------------------------- Stage 7: ctx = probs @ V ------------------------
// vT is [b][h*DH+d][LP]: B operand contiguous-in-K ([n][k] form).

__global__ void ctx_kernel(const float* __restrict__ p,
                           const float* __restrict__ vT,
                           float* __restrict__ ctx) {
  const int wave = threadIdx.x >> 5;
  const int nt = wave;                    // d tile 0..3 (DH=64)
  const int mt = blockIdx.y;              // l tile 0..23
  const int bh = blockIdx.z;
  const int b = bh / NH, h = bh % NH;
  const float* pb  = p + (size_t)bh * LP * LP;
  const float* vTb = vT + (b * H + h * DH) * LP;
  v8f acc = {};
  for (int k0 = 0; k0 < LP; k0 += 64) {
    v16h a0 = frag_a(pb, LP, mt * 16, k0, L);
    v16h b0 = frag_b_nk(vTb, LP, nt * 16, k0, DH);
    v16h a1 = frag_a(pb, LP, mt * 16, k0 + 32, L);
    v16h b1 = frag_b_nk(vTb, LP, nt * 16, k0 + 32, DH);
    acc = wmma32(a0, b0, acc);
    acc = wmma32(a1, b1, acc);
  }
  const int lane = threadIdx.x & 31;
  const int n = nt * 16 + (lane & 15);
  const int mo = (lane >> 4) * 8;
#pragma unroll
  for (int r2 = 0; r2 < 8; ++r2) {
    int m = mt * 16 + mo + r2;
    if (m < L) ctx[(b * L + m) * H + h * DH + n] = acc[r2];
  }
}

// -------------------------- Stage 9: LayerNorm ------------------------------

__global__ void layernorm_kernel(const float* __restrict__ x,
                                 const float* __restrict__ g,
                                 const float* __restrict__ bb,
                                 float* __restrict__ y) {
  __shared__ float red[8];
  const float* xr = x + blockIdx.x * H;
  float* yr = y + blockIdx.x * H;
  const int tid = threadIdx.x;            // 256
  float s = 0.0f;
  for (int j = tid; j < H; j += 256) s += xr[j];
  for (int off = 16; off; off >>= 1) s += __shfl_xor(s, off, 32);
  if ((tid & 31) == 0) red[tid >> 5] = s;
  __syncthreads();
  s = 0.0f;
  for (int w = 0; w < 8; ++w) s += red[w];
  const float mu = s / (float)H;
  __syncthreads();
  float var = 0.0f;
  for (int j = tid; j < H; j += 256) { float d = xr[j] - mu; var += d * d; }
  for (int off = 16; off; off >>= 1) var += __shfl_xor(var, off, 32);
  if ((tid & 31) == 0) red[tid >> 5] = var;
  __syncthreads();
  var = 0.0f;
  for (int w = 0; w < 8; ++w) var += red[w];
  const float inv = rsqrtf(var / (float)H + 1e-12f);
  for (int j = tid; j < H; j += 256) yr[j] = (xr[j] - mu) * inv * g[j] + bb[j];
}

// -------------------------- Stage 12: pairwise sel --------------------------
// sel[b,i,r,j] = sum_h tanh(a[b,j,h] + b_[b,i,h] + suv_b[h]) * rel_emb[r,h]
// A = rel_emb (M=R=12 in a 16-tile, clamped), B = tanh'd uv block generated
// on the fly (each uv element computed exactly once).

__device__ __forceinline__ void uv_half(v16h& bf, const float4* pa,
                                        const float4* pb, const float4* ps) {
#pragma unroll
  for (int q4 = 0; q4 < 4; ++q4) {
    float4 xa = pa[q4];
    float4 xb = pb[q4];
    float4 xs = ps[q4];
    bf[4 * q4 + 0] = (_Float16)tanhf(xa.x + xb.x + xs.x);
    bf[4 * q4 + 1] = (_Float16)tanhf(xa.y + xb.y + xs.y);
    bf[4 * q4 + 2] = (_Float16)tanhf(xa.z + xb.z + xs.z);
    bf[4 * q4 + 3] = (_Float16)tanhf(xa.w + xb.w + xs.w);
  }
}

__global__ void sel_kernel(const float* __restrict__ aM,
                           const float* __restrict__ bM,
                           const float* __restrict__ suv_b,
                           const float* __restrict__ rel,
                           float* __restrict__ sel) {
  const int wave = threadIdx.x >> 5;
  const int jt = blockIdx.x * 4 + wave;   // j tile 0..23
  const int i  = blockIdx.y;              // 0..L-1
  const int b  = blockIdx.z;
  const int lane = threadIdx.x & 31;
  const int n  = lane & 15;
  const int kb = (lane >> 4) * 16;
  const int j  = jt * 16 + n;
  const int jc = (j < L) ? j : (L - 1);   // clamp: dup cols never stored
  const float* bi = bM + (b * L + i) * H;
  const float* aj = aM + (b * L + jc) * H;
  v8f acc = {};
  for (int k0 = 0; k0 < H; k0 += 64) {
    v16h a0 = frag_a(rel, H, 0, k0, R);
    v16h a1 = frag_a(rel, H, 0, k0 + 32, R);
    v16h b0, b1;
    uv_half(b0, reinterpret_cast<const float4*>(aj + k0 + kb),
            reinterpret_cast<const float4*>(bi + k0 + kb),
            reinterpret_cast<const float4*>(suv_b + k0 + kb));
    uv_half(b1, reinterpret_cast<const float4*>(aj + k0 + 32 + kb),
            reinterpret_cast<const float4*>(bi + k0 + 32 + kb),
            reinterpret_cast<const float4*>(suv_b + k0 + 32 + kb));
    acc = wmma32(a0, b0, acc);
    acc = wmma32(a1, b1, acc);
  }
  const int mo = (lane >> 4) * 8;
  if (j < L) {
#pragma unroll
    for (int r2 = 0; r2 < 8; ++r2) {
      int r = mo + r2;
      if (r < R) sel[((b * L + i) * R + r) * L + j] = acc[r2];
    }
  }
}

// -------------------------- launch -----------------------------------------

extern "C" void kernel_launch(void* const* d_in, const int* in_sizes, int n_in,
                              void* d_out, int out_size, void* d_ws, size_t ws_size,
                              hipStream_t stream) {
  (void)in_sizes; (void)n_in; (void)out_size; (void)ws_size;
  const float* seq     = (const float*)d_in[0];
  const float* pool    = (const float*)d_in[1];
  const float* smask   = (const float*)d_in[2];
  const float* comb_W  = (const float*)d_in[3];
  const float* comb_b  = (const float*)d_in[4];
  const float* Wih     = (const float*)d_in[5];
  const float* Whh     = (const float*)d_in[6];
  const float* bih     = (const float*)d_in[7];
  const float* bhh     = (const float*)d_in[8];
  const float* Wq      = (const float*)d_in[9];
  const float* bq      = (const float*)d_in[10];
  const float* Wk      = (const float*)d_in[11];
  const float* bk      = (const float*)d_in[12];
  const float* Wv      = (const float*)d_in[13];
  const float* bv      = (const float*)d_in[14];
  const float* Wo      = (const float*)d_in[15];
  const float* bo      = (const float*)d_in[16];
  const float* ln_g    = (const float*)d_in[17];
  const float* ln_b    = (const float*)d_in[18];
  const float* cls_W   = (const float*)d_in[19];
  const float* cls_b   = (const float*)d_in[20];
  const float* rel_emb = (const float*)d_in[21];
  const float* su_W    = (const float*)d_in[22];
  const float* su_b    = (const float*)d_in[23];
  const float* sv_W    = (const float*)d_in[24];
  const float* sv_b    = (const float*)d_in[25];
  const float* suv_W   = (const float*)d_in[26];
  const float* suv_b   = (const float*)d_in[27];

  float* out = (float*)d_out;
  const size_t SEL_SZ = (size_t)B * L * R * L;
  float* classify = out + SEL_SZ;

  // workspace layout (floats; every offset a multiple of 16 floats = 64B)
  float* ws = (float*)d_ws;
  const size_t ML = (size_t)B * L;               // 738
  float* xin   = ws;  ws += ML * H2;
  float* xcomb = ws;  ws += ML * H;
  float* xl    = ws;  ws += ML * H;
  float* qb    = ws;  ws += ML * H;
  float* kbuf  = ws;  ws += ML * H;
  float* vT    = ws;  ws += (size_t)B * H * LP;  // transposed V, padded K
  float* sc    = ws;  ws += (size_t)B * NH * LP * LP;
  float* ctx   = ws;  ws += ML * H;
  float* preln = ws;  ws += ML * H;
  float* attn  = ws;  ws += ML * H;
  float* uB    = ws;  ws += ML * H;
  float* vB    = ws;  ws += ML * H;
  float* aMat  = ws;  ws += ML * H;
  float* bMat  = ws;  ws += ML * H;

  const int M = (int)ML;
  dim3 gemmBlk(128);
  auto gemmGrid = [](int N_, int M_) { return dim3((N_ + 63) / 64, (M_ + 15) / 16); };

  // 1. sliding max spans + pool concat
  spans_kernel<<<dim3(B * L), dim3(192), 0, stream>>>(seq, pool, xin);

  // 2. x = tanh(xin @ combine_W^T + b)
  gemm_nk_kernel<true, false, true><<<gemmGrid(H, M), gemmBlk, 0, stream>>>(
      xin, comb_W, comb_b, nullptr, xcomb, M, H, H2, H2, H2, H, 0);

  // 3. sequential LSTM (persistent workgroup, LDS-resident state)
  lstm_kernel<<<dim3(1), dim3(512), 0, stream>>>(xcomb, Wih, Whh, bih, bhh, xl);

  // 4. q/k projections; v stored transposed per head (K-contiguous for P@V)
  gemm_nk_kernel<true, false, false><<<gemmGrid(H, M), gemmBlk, 0, stream>>>(
      xl, Wq, bq, nullptr, qb, M, H, H, H, H, H, 0);
  gemm_nk_kernel<true, false, false><<<gemmGrid(H, M), gemmBlk, 0, stream>>>(
      xl, Wk, bk, nullptr, kbuf, M, H, H, H, H, H, 0);
  gemm_nk_t_kernel<<<gemmGrid(H, M), gemmBlk, 0, stream>>>(
      xl, Wv, bv, vT, M, H, H, H, H);
  vt_padzero_kernel<<<dim3((B * H * (LP - L) + 255) / 256), dim3(256), 0, stream>>>(vT);

  // 5. scores (padded, zero-filled pads)  6. softmax  7. ctx = P V
  scores_kernel<<<dim3(6, 24, B * NH), dim3(128), 0, stream>>>(qb, kbuf, smask, sc);
  softmax_kernel<<<dim3(B * NH * L), dim3(128), 0, stream>>>(sc);
  ctx_kernel<<<dim3(1, 24, B * NH), dim3(128), 0, stream>>>(sc, vT, ctx);

  // 8. preln = ctx @ Wo^T + bo + x   9. layernorm
  gemm_nk_kernel<true, true, false><<<gemmGrid(H, M), gemmBlk, 0, stream>>>(
      ctx, Wo, bo, xl, preln, M, H, H, H, H, H, H);
  layernorm_kernel<<<dim3(M), dim3(256), 0, stream>>>(preln, ln_g, ln_b, attn);

  // 10. classify (N=2)
  gemm_nk_kernel<true, false, false><<<gemmGrid(TAGS, M), gemmBlk, 0, stream>>>(
      attn, cls_W, cls_b, nullptr, classify, M, TAGS, H, H, H, TAGS, 0);

  // 11. u/v tanh projections, then a / b_ halves of suv_W
  gemm_nk_kernel<true, false, true><<<gemmGrid(H, M), gemmBlk, 0, stream>>>(
      attn, su_W, su_b, nullptr, uB, M, H, H, H, H, H, 0);
  gemm_nk_kernel<true, false, true><<<gemmGrid(H, M), gemmBlk, 0, stream>>>(
      attn, sv_W, sv_b, nullptr, vB, M, H, H, H, H, H, 0);
  gemm_nk_kernel<false, false, false><<<gemmGrid(H, M), gemmBlk, 0, stream>>>(
      uB, suv_W, nullptr, nullptr, aMat, M, H, H, H, H2, H, 0);
  gemm_nk_kernel<false, false, false><<<gemmGrid(H, M), gemmBlk, 0, stream>>>(
      vB, suv_W + H, nullptr, nullptr, bMat, M, H, H, H, H2, H, 0);

  // 12. sel = rel_emb @ tanh(a_j + b_i + bias) per (b, i)
  sel_kernel<<<dim3(6, L, B), dim3(128), 0, stream>>>(aMat, bMat, suv_b, rel_emb, out);
}
